// CellularTransformer_67345087201410
// MI455X (gfx1250) — matrix operand order repaired and checked by reading, not compiled, not run on hardware
//
#include <hip/hip_runtime.h>
#include <math.h>

// ---------------------------------------------------------------------------
// CDNA5 (gfx1250) wave32 WMMA cellular transformer.
//  * GEMMs: LDS-free v_wmma_f32_16x16x32_bf16; weights pre-transposed to
//    [N][K] bf16 so BOTH fragments are contiguous 16B global loads.
//  * Attention: flash-style; K/V staged with global_load_async_to_lds_b32.
// ---------------------------------------------------------------------------

typedef __attribute__((ext_vector_type(16))) __bf16 v16bf;
typedef __attribute__((ext_vector_type(8)))  float  v8f;

union BFrag { v16bf v; unsigned short s[16]; uint4 q[2]; };

__device__ __forceinline__ unsigned short f2bf(float f) {
  union { float f; unsigned u; } x; x.f = f;
  unsigned u = x.u + 0x7FFFu + ((x.u >> 16) & 1u);   // round-to-nearest-even
  return (unsigned short)(u >> 16);
}

#define GF_ACC     1
#define GF_RELU    2
#define GF_BF16OUT 4

// -------------------------------------------------------------------- utils
__global__ void zero_kernel(float* p, int n) {
  for (int i = blockIdx.x * blockDim.x + threadIdx.x; i < n; i += gridDim.x * blockDim.x)
    p[i] = 0.f;
}

// dst[b][n][k] (row stride `stride`, col offset `off`) = bf16(src[b][k][n])
__global__ void cvt_t_kernel(const float* __restrict__ src, unsigned short* __restrict__ dst,
                             int K, int N, int stride, int off, int total) {
  int KN = K * N;
  for (int i = blockIdx.x * blockDim.x + threadIdx.x; i < total; i += gridDim.x * blockDim.x) {
    int b = i / KN, r = i - b * KN;
    int k = r / N, n = r - k * N;
    dst[(size_t)b * N * stride + (size_t)n * stride + off + k] = f2bf(src[i]);
  }
}

// emb activation pack: [x(32) | pe(16) | zeros(16)] per row, bf16
__global__ void pack_emb_kernel(const float* __restrict__ x, const float* __restrict__ pe,
                                unsigned short* __restrict__ out, int rows) {
  int total = rows * 64;
  for (int i = blockIdx.x * blockDim.x + threadIdx.x; i < total; i += gridDim.x * blockDim.x) {
    int row = i >> 6, c = i & 63;
    float v = (c < 32) ? x[row * 32 + c] : (c < 48 ? pe[row * 16 + (c - 32)] : 0.f);
    out[i] = f2bf(v);
  }
}

// ------------------------------------------------------------ generic GEMM
// out[M,N] = Act[M,K](bf16) @ W[K,N] (+bias)(+C_old)(relu); W given as Wt[N][K].
// D tile computed transposed: A = Wt rows, B = Act rows. LDS-free.
// 8 waves/block; wave owns a 16-row M strip, 64 N-cols (4 accumulators).
__global__ __launch_bounds__(256)
void gemm_wmma(const unsigned short* __restrict__ Act, int lda,
               const unsigned short* __restrict__ Wt, int ldw,
               const float* __restrict__ bias,
               float* __restrict__ C, unsigned short* __restrict__ Cb, int ldc,
               int K, int flags)
{
  const int lane = threadIdx.x & 31;
  const int wv   = threadIdx.x >> 5;
  const int Mbase = blockIdx.x * 128 + wv * 16;
  const int tn    = blockIdx.y * 64;
  const int m  = lane & 15;
  const int hi = (lane >> 4) << 3;   // +8 K offset, upper half-wave (unused here)
  const int kb = (lane >> 4) << 4;   // +16 K offset, upper half-wave (B frag)
  const unsigned short* actRow = Act + (size_t)(Mbase + m) * lda + kb;
  v8f cacc[4] = {{}, {}, {}, {}};

  for (int k0 = 0; k0 < K; k0 += 32) {
    BFrag b;                                        // activations, k = kb+e
    b.q[0] = *(const uint4*)(actRow + k0);
    b.q[1] = *(const uint4*)(actRow + k0 + 8);
    if (k0 + 32 < K) __builtin_prefetch(actRow + k0 + 32, 0, 3);
    #pragma unroll
    for (int nt = 0; nt < 4; ++nt) {
      const unsigned short* wp = Wt + (size_t)(tn + nt * 16 + m) * ldw + k0 + kb;
      BFrag a;                                      // weights, k = kb+e
      a.q[0] = *(const uint4*)(wp);
      a.q[1] = *(const uint4*)(wp + 8);
      cacc[nt] = __builtin_amdgcn_wmma_f32_16x16x32_bf16(false, a.v, false, b.v,
                                                         (short)0, cacc[nt], false, false);
    }
  }

  const int rb   = (lane >> 4) << 3;
  const int mrow = Mbase + m;
  #pragma unroll
  for (int nt = 0; nt < 4; ++nt) {
    #pragma unroll
    for (int r = 0; r < 8; ++r) {                   // contiguous 8-col run
      int gn = tn + nt * 16 + rb + r;
      float v = cacc[nt][r];
      if (bias) v += bias[gn];
      if (flags & GF_ACC) v += C[(size_t)mrow * ldc + gn];
      if (flags & GF_RELU) v = fmaxf(v, 0.f);
      if (C) C[(size_t)mrow * ldc + gn] = v;
      if (flags & GF_BF16OUT) Cb[(size_t)mrow * ldc + gn] = f2bf(v);
    }
  }
}

// ------------------------------------------------------- masked flash attn
// One wave per (head, 16-target tile); streams 32-source chunks.
// Q/K/V are bf16 [N][64]; K/V chunks staged via async-to-LDS copies.
__global__ __launch_bounds__(32)
void attn_kernel(const unsigned short* __restrict__ Qbf, const unsigned short* __restrict__ Kbf,
                 const unsigned short* __restrict__ Vbf, unsigned short* __restrict__ Ob,
                 const float* __restrict__ maskM, int mld, int mtrans,
                 const int* __restrict__ belT, const int* __restrict__ belS,
                 int Nt, int Ns, float scale)
{
  __shared__ __align__(16) unsigned short Ks[32 * 8];
  __shared__ __align__(16) unsigned short Vs[32 * 8];
  __shared__ __align__(16) unsigned short Ps[16 * 32];
  __shared__ float Ssc[16 * 32];
  __shared__ float Olds[16 * 8];
  __shared__ float mrow[16], lrow[16], arow[16];

  const int lane = threadIdx.x;
  const int head = blockIdx.y;
  const int tile = blockIdx.x * 16;
  const int m  = lane & 15;
  const int hi = (lane >> 4) << 3;
  const int kb = (lane >> 4) << 4;
  const int rb = (lane >> 4) << 3;
  const float NEG = -__builtin_inff();

  BFrag aq;                                          // Q 16x8 zero-padded to 16x32
  #pragma unroll
  for (int e = 0; e < 16; ++e) aq.s[e] = 0;
  if (lane < 16)
    aq.q[0] = *(const uint4*)&Qbf[(size_t)(tile + m) * 64 + head * 8];

  for (int t = lane; t < 128; t += 32) Olds[t] = 0.f;
  if (lane < 16) { mrow[lane] = -1e30f; lrow[lane] = 0.f; }
  __syncthreads();

  for (int s0 = 0; s0 < Ns; s0 += 32) {
    // ---- async DMA stage of K/V chunk (raw bf16 copy, ASYNCcnt path)
    for (int t = lane; t < 128; t += 32) {
      int row = t >> 2, w = (t & 3) * 2;             // 4 dwords per 8-elem row
      unsigned ldsK = (unsigned)(size_t)&Ks[row * 8 + w];
      const unsigned short* gK = &Kbf[(size_t)(s0 + row) * 64 + head * 8 + w];
      asm volatile("global_load_async_to_lds_b32 %0, %1, off" :: "v"(ldsK), "v"(gK) : "memory");
      unsigned ldsV = (unsigned)(size_t)&Vs[row * 8 + w];
      const unsigned short* gV = &Vbf[(size_t)(s0 + row) * 64 + head * 8 + w];
      asm volatile("global_load_async_to_lds_b32 %0, %1, off" :: "v"(ldsV), "v"(gV) : "memory");
    }
    asm volatile("s_wait_asynccnt 0" ::: "memory");
    __syncthreads();

    #pragma unroll
    for (int hh = 0; hh < 2; ++hh) {                 // two 16-source halves
      BFrag bk;                                      // B = K^T (8x16, K padded)
      #pragma unroll
      for (int e = 0; e < 16; ++e) bk.s[e] = 0;
      if (lane < 16)
        bk.q[0] = *(const uint4*)&Ks[(hh * 16 + m) * 8];
      v8f cz = {};
      v8f s = __builtin_amdgcn_wmma_f32_16x16x32_bf16(false, aq.v, false, bk.v, (short)0, cz, false, false);
      #pragma unroll
      for (int r = 0; r < 8; ++r) {                  // mask + scale -> LDS
        int row = rb + r, gt = tile + row, gs = s0 + hh * 16 + m;
        float val = NEG;
        {
          float mv = mtrans ? maskM[(size_t)gs * mld + gt] : maskM[(size_t)gt * mld + gs];
          if (mv > 0.f && belT[gt] == belS[gs]) val = s[r] * scale;
        }
        Ssc[row * 32 + hh * 16 + m] = val;
      }
    }
    __syncthreads();

    if (lane < 16) {                                 // online softmax row stats
      int row = lane;
      float cm = NEG;
      #pragma unroll
      for (int j = 0; j < 32; ++j) cm = fmaxf(cm, Ssc[row * 32 + j]);
      float mo = mrow[row], mn = fmaxf(mo, cm);      // stays finite (>= -1e30)
      float al = __expf(mo - mn), sum = 0.f;
      #pragma unroll
      for (int j = 0; j < 32; ++j) {
        float p = __expf(Ssc[row * 32 + j] - mn);    // -inf -> 0 (masked)
        Ps[row * 32 + j] = f2bf(p);
        sum += p;
      }
      lrow[row] = lrow[row] * al + sum;
      mrow[row] = mn;
      arow[row] = al;
    }
    __syncthreads();

    BFrag ap, bv;                                    // O += P(16x32) @ V(32x8)
    ap.q[0] = *(const uint4*)&Ps[m * 32 + hi];       // kk = e+hi
    ap.q[1] = *(const uint4*)&Ps[m * 32 + 16 + hi];  // kk = 16+hi+(e-8)
    #pragma unroll
    for (int e = 0; e < 16; ++e)
      bv.s[e] = (m < 8) ? Vs[(hi + e + (e & 8)) * 8 + m] : (unsigned short)0;
    v8f cz = {};
    v8f oc = __builtin_amdgcn_wmma_f32_16x16x32_bf16(false, ap.v, false, bv.v, (short)0, cz, false, false);
    if (m < 8) {
      #pragma unroll
      for (int r = 0; r < 8; ++r) {
        int row = rb + r;
        Olds[row * 8 + m] = Olds[row * 8 + m] * arow[row] + oc[r];
      }
    }
    __syncthreads();
  }

  for (int t = lane; t < 128; t += 32) {
    int row = t >> 3, d = t & 7, gt = tile + row;
    float l = lrow[row];
    Ob[(size_t)gt * 64 + head * 8 + d] = f2bf(l > 0.f ? Olds[t] / l : 0.f);
  }
}

// -------------------------------------------------------------- layernorm
// out = LN(a+b; g,be) (f32) with optional bf16 mirror.
__global__ __launch_bounds__(32)
void ln_add_kernel(const float* __restrict__ a, const float* __restrict__ b,
                   const float* __restrict__ g, const float* __restrict__ be,
                   float* __restrict__ out, unsigned short* __restrict__ outb)
{
  int row = blockIdx.x, lane = threadIdx.x;
  float v0 = a[row * 64 + lane]      + b[row * 64 + lane];
  float v1 = a[row * 64 + lane + 32] + b[row * 64 + lane + 32];
  float s = v0 + v1;
  #pragma unroll
  for (int o = 16; o; o >>= 1) s += __shfl_xor(s, o, 32);
  float mu = s * (1.f / 64.f);
  float d0 = v0 - mu, d1 = v1 - mu;
  float vs = d0 * d0 + d1 * d1;
  #pragma unroll
  for (int o = 16; o; o >>= 1) vs += __shfl_xor(vs, o, 32);
  float rs = rsqrtf(vs * (1.f / 64.f) + 1e-5f);
  float r0 = d0 * rs * g[lane]      + be[lane];
  float r1 = d1 * rs * g[lane + 32] + be[lane + 32];
  out[row * 64 + lane]      = r0;
  out[row * 64 + lane + 32] = r1;
  if (outb) {
    outb[row * 64 + lane]      = f2bf(r0);
    outb[row * 64 + lane + 32] = f2bf(r1);
  }
}

// ------------------------------------------------------------- pooling/head
__global__ void pool_kernel(const float* __restrict__ hs,
                            const int* __restrict__ bel0, const int* __restrict__ bel1,
                            const int* __restrict__ bel2,
                            float* __restrict__ pooled, float* __restrict__ cnt)
{
  int total = 6144 * 64;
  for (int i = blockIdx.x * blockDim.x + threadIdx.x; i < total; i += gridDim.x * blockDim.x) {
    int row = i >> 6, col = i & 63;
    int b = (row < 1536) ? bel0[row] : (row < 4608) ? bel1[row - 1536] : bel2[row - 4608];
    atomicAdd(&pooled[b * 64 + col], hs[i]);
    if (col == 0) atomicAdd(&cnt[b], 1.f);
  }
}

__global__ __launch_bounds__(512)
void head_kernel(const float* __restrict__ pooled, const float* __restrict__ cnt,
                 const float* __restrict__ Wh1, const float* __restrict__ bh1,
                 const float* __restrict__ Wh2, const float* __restrict__ bh2,
                 const float* __restrict__ Wh3, const float* __restrict__ bh3,
                 float* __restrict__ out)
{
  __shared__ float p[512], y1[512], y2[512];
  int tid = threadIdx.x;
  {
    int bi = tid >> 6;
    p[tid] = pooled[tid] / fmaxf(cnt[bi], 1.f);
  }
  __syncthreads();
  {
    int bi = tid >> 6, j = tid & 63;
    float s = bh1[j];
    for (int k = 0; k < 64; ++k) s += p[bi * 64 + k] * Wh1[k * 64 + j];
    y1[tid] = fmaxf(s, 0.f);
  }
  __syncthreads();
  {
    int bi = tid >> 6, j = tid & 63;
    float s = bh2[j];
    for (int k = 0; k < 64; ++k) s += y1[bi * 64 + k] * Wh2[k * 64 + j];
    y2[tid] = fmaxf(s, 0.f);
  }
  __syncthreads();
  if (tid < 80) {
    int bi = tid / 10, j = tid % 10;
    float s = bh3[j];
    for (int k = 0; k < 64; ++k) s += y2[bi * 64 + k] * Wh3[k * 10 + j];
    out[tid] = s;
  }
}

// ---------------------------------------------------------------- driver
extern "C" void kernel_launch(void* const* d_in, const int* in_sizes, int n_in,
                              void* d_out, int out_size, void* d_ws, size_t ws_size,
                              hipStream_t stream)
{
  (void)in_sizes; (void)n_in; (void)out_size; (void)ws_size;
  const int Nd[3] = {1536, 3072, 1536};
  const int HSoff[3] = {0, 1536 * 64, 4608 * 64};
  const int L = 2;

  const float* x[3]  = {(const float*)d_in[0], (const float*)d_in[3], (const float*)d_in[6]};
  const float* pe[3] = {(const float*)d_in[1], (const float*)d_in[4], (const float*)d_in[7]};
  const int*  bel[3] = {(const int*)d_in[2], (const int*)d_in[5], (const int*)d_in[8]};
  const float* adj00 = (const float*)d_in[9];
  const float* adj11 = (const float*)d_in[10];
  const float* adj22 = (const float*)d_in[11];
  const float* b01   = (const float*)d_in[12];
  const float* b12   = (const float*)d_in[13];
  const float* Wf[3] = {(const float*)d_in[14], (const float*)d_in[17], (const float*)d_in[20]};
  const float* bfb[3]= {(const float*)d_in[15], (const float*)d_in[18], (const float*)d_in[21]};
  const float* Wp[3] = {(const float*)d_in[16], (const float*)d_in[19], (const float*)d_in[22]};
  const float* Wq = (const float*)d_in[23]; const float* bq = (const float*)d_in[24];
  const float* Wk = (const float*)d_in[25]; const float* bk = (const float*)d_in[26];
  const float* Wv = (const float*)d_in[27]; const float* bv = (const float*)d_in[28];
  const float* Wo = (const float*)d_in[29]; const float* bo = (const float*)d_in[30];
  const float* g1 = (const float*)d_in[31]; const float* be1 = (const float*)d_in[32];
  const float* g2 = (const float*)d_in[33]; const float* be2 = (const float*)d_in[34];
  const float* Wff1 = (const float*)d_in[35]; const float* bff1 = (const float*)d_in[36];
  const float* Wff2 = (const float*)d_in[37]; const float* bff2 = (const float*)d_in[38];
  const float* Wh1 = (const float*)d_in[39]; const float* bh1 = (const float*)d_in[40];
  const float* Wh2 = (const float*)d_in[41]; const float* bh2 = (const float*)d_in[42];
  const float* Wh3 = (const float*)d_in[43]; const float* bh3 = (const float*)d_in[44];

  // ---- workspace: f32 region, then bf16 (ushort) region
  float* wsf = (float*)d_ws;
  float* hs   = wsf;                      // 6144*64
  float* acc  = hs  + 393216;
  float* hb   = acc + 393216;
  float* ffo  = hb  + 393216;             // 3072*64
  float* pooled = ffo + 196608;           // 512
  float* pcnt   = pooled + 512;           // 8 (32 B -> keeps 16B alignment)
  unsigned short* us = (unsigned short*)(pcnt + 8);
  unsigned short* embA  = us;             // 6144*64 = 393216
  unsigned short* hs_bf = embA  + 393216;
  unsigned short* hb_bf = hs_bf + 393216;
  unsigned short* ffb_bf= hb_bf + 393216; // 3072*256 = 786432
  unsigned short* Qbf   = ffb_bf + 786432;// 3072*64 = 196608
  unsigned short* Kbf   = Qbf + 196608;
  unsigned short* Vbf   = Kbf + 196608;
  unsigned short* Obf   = Vbf + 196608;
  unsigned short* WqT   = Obf + 196608;   // 14*64*64 = 57344 each
  unsigned short* WkT   = WqT + 57344;
  unsigned short* WvT   = WkT + 57344;
  unsigned short* WoT   = WvT + 57344;
  unsigned short* W1T   = WoT + 57344;    // 6*256*64 = 98304
  unsigned short* W2T   = W1T + 98304;    // 6*64*256 = 98304
  unsigned short* WembT = W2T + 98304;    // 3*64*64 = 12288 (zero-padded K)

  auto cvtT = [&](const float* s, unsigned short* d, int K, int N, int stride, int off, int batch) {
    int total = batch * K * N;
    cvt_t_kernel<<<(total + 255) / 256, 256, 0, stream>>>(s, d, K, N, stride, off, total);
  };
  // transposed bf16 weights
  cvtT(Wq, WqT, 64, 64, 64, 0, 14);
  cvtT(Wk, WkT, 64, 64, 64, 0, 14);
  cvtT(Wv, WvT, 64, 64, 64, 0, 14);
  cvtT(Wo, WoT, 64, 64, 64, 0, 14);
  cvtT(Wff1, W1T, 64, 256, 64, 0, 6);
  cvtT(Wff2, W2T, 256, 64, 256, 0, 6);
  zero_kernel<<<(6144 + 255) / 256, 256, 0, stream>>>((float*)WembT, 6144); // 12288 ushorts
  for (int d = 0; d < 3; ++d) {
    cvtT(Wf[d], WembT + d * 4096, 32, 64, 64, 0, 1);
    cvtT(Wp[d], WembT + d * 4096, 16, 64, 64, 32, 1);
    pack_emb_kernel<<<(Nd[d] * 64 + 255) / 256, 256, 0, stream>>>(x[d], pe[d], embA + HSoff[d], Nd[d]);
  }

  auto gemm = [&](const unsigned short* Act, int lda, const unsigned short* Wt, int ldw,
                  const float* bias, float* C, unsigned short* Cb, int ldc,
                  int M, int N, int K, int flags) {
    dim3 grid(M / 128, N / 64);
    gemm_wmma<<<grid, 256, 0, stream>>>(Act, lda, Wt, ldw, bias, C, Cb, ldc, K, flags);
  };

  // embed: hs[d] = [x|pe|0] @ [Wf;Wp;0] + bf    (single K=64 GEMM, f32+bf16 out)
  for (int d = 0; d < 3; ++d)
    gemm(embA + HSoff[d], 64, WembT + d * 4096, 64, bfb[d],
         hs + HSoff[d], hs_bf + HSoff[d], 64, Nd[d], 64, 64, GF_BF16OUT);

  const int sdA[7] = {0, 0, 1, 1, 1, 2, 2};
  const int tdA[7] = {0, 1, 0, 1, 2, 1, 2};
  const float* mp[7] = {adj00, b01, b01, adj11, b12, b12, adj22};
  const int mld[7] = {1536, 3072, 3072, 3072, 1536, 1536, 1536};
  const int mtr[7] = {0, 1, 0, 0, 1, 0, 0};
  const float scale = 0.35355339059327373f;  // 1/sqrt(8)

  for (int l = 0; l < L; ++l) {
    zero_kernel<<<(393216 + 255) / 256, 256, 0, stream>>>(acc, 393216);
    for (int i = 0; i < 7; ++i) {
      int sd = sdA[i], td = tdA[i];
      int Nt = Nd[td], Ns = Nd[sd];
      int wqo = (l * 7 + i) * 4096, bqo = (l * 7 + i) * 64;
      gemm(hs_bf + HSoff[td], 64, WqT + wqo, 64, bq + bqo, nullptr, Qbf, 64, Nt, 64, 64, GF_BF16OUT);
      gemm(hs_bf + HSoff[sd], 64, WkT + wqo, 64, bk + bqo, nullptr, Kbf, 64, Ns, 64, 64, GF_BF16OUT);
      gemm(hs_bf + HSoff[sd], 64, WvT + wqo, 64, bv + bqo, nullptr, Vbf, 64, Ns, 64, 64, GF_BF16OUT);
      dim3 ag(Nt / 16, 8);
      attn_kernel<<<ag, 32, 0, stream>>>(Qbf, Kbf, Vbf, Obf, mp[i], mld[i], mtr[i],
                                         bel[td], bel[sd], Nt, Ns, scale);
      gemm(Obf, 64, WoT + wqo, 64, bo + bqo, acc + HSoff[td], nullptr, 64, Nt, 64, 64, GF_ACC);
    }
    for (int d = 0; d < 3; ++d) {
      int gdo = (l * 3 + d) * 64;
      ln_add_kernel<<<Nd[d], 32, 0, stream>>>(hs + HSoff[d], acc + HSoff[d],
                                              g1 + gdo, be1 + gdo, hb + HSoff[d], hb_bf + HSoff[d]);
      int w1o = (l * 3 + d) * 16384, b1o = (l * 3 + d) * 256;
      gemm(hb_bf + HSoff[d], 64, W1T + w1o, 64, bff1 + b1o, nullptr, ffb_bf, 256,
           Nd[d], 256, 64, GF_RELU | GF_BF16OUT);
      gemm(ffb_bf, 256, W2T + w1o, 256, bff2 + gdo, ffo, nullptr, 64, Nd[d], 64, 256, 0);
      ln_add_kernel<<<Nd[d], 32, 0, stream>>>(hb + HSoff[d], ffo,
                                              g2 + gdo, be2 + gdo, hs + HSoff[d], hs_bf + HSoff[d]);
    }
  }

  zero_kernel<<<3, 256, 0, stream>>>(pooled, 520);
  pool_kernel<<<512, 256, 0, stream>>>(hs, bel[0], bel[1], bel[2], pooled, pcnt);
  head_kernel<<<1, 512, 0, stream>>>(pooled, pcnt, Wh1, bh1, Wh2, bh2, Wh3, bh3, (float*)d_out);
}
